// GCNEncoder_22308060136294
// MI455X (gfx1250) — matrix operand order repaired and checked by reading, not compiled, not run on hardware
//
#include <hip/hip_runtime.h>

typedef __attribute__((ext_vector_type(16))) __bf16 v16bf;
typedef __attribute__((ext_vector_type(8)))  float  v8f;

#define N_NODES 100000
#define E_EDGES 1600000
#define F_IN    128
#define F_HID   128
#define F_OUT   64

// ---------------------------------------------------------------- small utils

__global__ void init_deg_kernel(float* deg, int n) {
    int i = blockIdx.x * blockDim.x + threadIdx.x;
    if (i < n) deg[i] = 1.0f;   // self-loop contributes 1 to every node's degree
}

__global__ void count_deg_kernel(const int* __restrict__ col, float* deg, int e) {
    int i = blockIdx.x * blockDim.x + threadIdx.x;
    if (i < e) unsafeAtomicAdd(&deg[col[i]], 1.0f);
}

__global__ void dinv_kernel(const float* __restrict__ deg, float* dinv, int n) {
    int i = blockIdx.x * blockDim.x + threadIdx.x;
    if (i < n) {
        float d = deg[i];
        dinv[i] = (d > 0.0f) ? rsqrtf(d) : 0.0f;
    }
}

__global__ void zero_kernel(float* p, long total) {
    long i = (long)blockIdx.x * blockDim.x + threadIdx.x;
    if (i < total) p[i] = 0.0f;
}

// ---------------------------------------------------------------- WMMA GEMM
// H[nrows x F] = X[nrows x 128] * W[128 x F], bf16 inputs, f32 accumulate.
// 128 threads = 4 waves; block covers 64 rows; wave w covers rows [16w,16w+16).
// LDS strides are 136 elements (272 B): 16B-aligned rows for ds_load_b128 and
// 4-DWORD bank step per row so 16 lanes' b128 loads tile all 64 banks.
// nrows must be a multiple of 16 (true here: 100000), so each wave's 16-row
// strip is uniformly valid or invalid -> scalar-branch store epilogue.
template <int F>
__global__ __launch_bounds__(128)
void gemm_bf16_wmma_kernel(const float* __restrict__ X,
                           const float* __restrict__ W,
                           float* __restrict__ H, int nrows) {
    constexpr int K  = 128;
    constexpr int XS = 136;               // X tile LDS row stride (elements)
    constexpr int WS = 136;               // W^T LDS row stride (elements)
    __shared__ __bf16 sX[64 * XS];
    __shared__ __bf16 sWt[F * WS];        // sWt[n*WS + k] = W[k*F + n]

    const int t    = threadIdx.x;
    const int lane = t & 31;
    const int wave = t >> 5;
    const int rowBase = blockIdx.x * 64;

    // Stage X tile (64 x 128) -> bf16 LDS via float4 global loads; zero-pad OOB.
    for (int i = t; i < 64 * (K / 4); i += 128) {
        int r  = i >> 5;          // / (K/4)
        int c4 = i & 31;
        int gr = rowBase + r;
        float4 v = make_float4(0.f, 0.f, 0.f, 0.f);
        if (gr < nrows) v = *(const float4*)(X + (long)gr * K + c4 * 4);
        __bf16* d = &sX[r * XS + c4 * 4];
        d[0] = (__bf16)v.x; d[1] = (__bf16)v.y; d[2] = (__bf16)v.z; d[3] = (__bf16)v.w;
    }
    // Stage W (K x F) transposed -> sWt[n][k]; consecutive threads read
    // consecutive n (coalesced global), scatter-write columns into LDS.
    for (int i = t; i < K * F; i += 128) {
        int k = i / F, n = i % F;
        sWt[n * WS + k] = (__bf16)W[i];
    }
    __syncthreads();

    union Frag { v16bf v; uint4 q[2]; };

    const int half = lane >> 4;   // wave32 half-select per ISA VGPR layouts
    const int m    = lane & 15;

    // Wave-uniform validity of this wave's 16-row strip (nrows % 16 == 0).
    const bool stripValid = (rowBase + wave * 16 + 16) <= nrows;

    // A fragments (nt-invariant): e=0..7 at K=kb+8*half+e, e=8..15 at +16.
    const __bf16* aRow = &sX[(wave * 16 + m) * XS + half * 8];
    Frag A[K / 32];
#pragma unroll
    for (int kt = 0; kt < K / 32; ++kt) {
        A[kt].q[0] = *(const uint4*)(aRow + kt * 32);
        A[kt].q[1] = *(const uint4*)(aRow + kt * 32 + 16);
    }

    for (int nt = 0; nt < F / 16; ++nt) {
        // B fragment: e=0..7 at K=kb+8*half+e, e=8..15 at +16, in sWt row n.
        const __bf16* bCol = &sWt[(nt * 16 + m) * WS + half * 8];
        v8f acc = {};
#pragma unroll
        for (int kt = 0; kt < K / 32; ++kt) {
            Frag B;
            B.q[0] = *(const uint4*)(bCol + kt * 32);
            B.q[1] = *(const uint4*)(bCol + kt * 32 + 16);
            acc = __builtin_amdgcn_wmma_f32_16x16x32_bf16(
                false, A[kt].v, false, B.v, (short)0, acc, false, false);
        }
        // D (16x16 f32): VGPR r -> row M = r + 8*half, col N = m
        if (stripValid) {
            float* hBase = H + (long)(rowBase + wave * 16 + (half << 3)) * F
                             + nt * 16 + m;
#pragma unroll
            for (int r = 0; r < 8; ++r) {
                hBase[(long)r * F] = acc[r];
            }
        }
    }
}

// ---------------------------------------------------------------- edge scatter
// agg[col[e]] += H[row[e]] * dinv[row]*dinv[col]; one thread per (edge, 4 feats)
template <int F>
__global__ void scatter_kernel(const float* __restrict__ H,
                               const int* __restrict__ row,
                               const int* __restrict__ col,
                               const float* __restrict__ dinv,
                               float* agg, int e) {
    constexpr int PF = F / 4;
    long w = (long)blockIdx.x * blockDim.x + threadIdx.x;
    long total = (long)e * PF;
    if (w >= total) return;
    int edge = (int)(w / PF);
    int j    = (int)(w % PF);
    int r = row[edge], c = col[edge];
    float norm = dinv[r] * dinv[c];
    const float4 h4 = *(const float4*)(H + (long)r * F + j * 4);
    float* dst = agg + (long)c * F + j * 4;
    unsafeAtomicAdd(dst + 0, h4.x * norm);
    unsafeAtomicAdd(dst + 1, h4.y * norm);
    unsafeAtomicAdd(dst + 2, h4.z * norm);
    unsafeAtomicAdd(dst + 3, h4.w * norm);
}

// ---------------------------------------------------------------- finalize
// agg[i,f] = (relu?)(agg[i,f] + H[i,f]*dinv[i]^2 + bias[f])
template <int F, bool RELU>
__global__ void finalize_kernel(float* __restrict__ agg,
                                const float* __restrict__ H,
                                const float* __restrict__ dinv,
                                const float* __restrict__ bias, int n) {
    long i = (long)blockIdx.x * blockDim.x + threadIdx.x;
    long total = (long)n * F;
    if (i >= total) return;
    int node = (int)(i / F);
    int f    = (int)(i % F);
    float d = dinv[node];
    float v = agg[i] + H[i] * d * d + bias[f];
    if (RELU) v = fmaxf(v, 0.0f);
    agg[i] = v;
}

// ---------------------------------------------------------------- launch

extern "C" void kernel_launch(void* const* d_in, const int* in_sizes, int n_in,
                              void* d_out, int out_size, void* d_ws, size_t ws_size,
                              hipStream_t stream) {
    const float* x  = (const float*)d_in[0];
    const int*   ei = (const int*)d_in[1];
    const float* W1 = (const float*)d_in[2];
    const float* b1 = (const float*)d_in[3];
    const float* W2 = (const float*)d_in[4];
    const float* b2 = (const float*)d_in[5];
    float* out = (float*)d_out;

    const int* row = ei;            // edge_index[0]
    const int* col = ei + E_EDGES;  // edge_index[1]

    float* ws   = (float*)d_ws;
    float* deg  = ws;                            // N
    float* dinv = deg + N_NODES;                 // N
    float* H1   = dinv + N_NODES;                // N * 128
    float* A1   = H1 + (long)N_NODES * F_HID;    // N * 128 (agg, then relu'd acts)
    float* H2   = A1 + (long)N_NODES * F_HID;    // N * 64

    const int T = 256;
    auto cdiv = [](long a, long b) { return (int)((a + b - 1) / b); };

    // degrees & normalization
    init_deg_kernel<<<cdiv(N_NODES, T), T, 0, stream>>>(deg, N_NODES);
    count_deg_kernel<<<cdiv(E_EDGES, T), T, 0, stream>>>(col, deg, E_EDGES);
    dinv_kernel<<<cdiv(N_NODES, T), T, 0, stream>>>(deg, dinv, N_NODES);

    // ---- layer 1: H1 = X @ W1 ; A1 = relu(scatter(H1) + selfloop + b1)
    gemm_bf16_wmma_kernel<F_HID><<<cdiv(N_NODES, 64), 128, 0, stream>>>(x, W1, H1, N_NODES);
    zero_kernel<<<cdiv((long)N_NODES * F_HID, T), T, 0, stream>>>(A1, (long)N_NODES * F_HID);
    scatter_kernel<F_HID><<<cdiv((long)E_EDGES * (F_HID / 4), T), T, 0, stream>>>(
        H1, row, col, dinv, A1, E_EDGES);
    finalize_kernel<F_HID, true><<<cdiv((long)N_NODES * F_HID, T), T, 0, stream>>>(
        A1, H1, dinv, b1, N_NODES);

    // ---- layer 2: H2 = A1 @ W2 ; out = scatter(H2) + selfloop + b2
    gemm_bf16_wmma_kernel<F_OUT><<<cdiv(N_NODES, 64), 128, 0, stream>>>(A1, W2, H2, N_NODES);
    zero_kernel<<<cdiv((long)N_NODES * F_OUT, T), T, 0, stream>>>(out, (long)N_NODES * F_OUT);
    scatter_kernel<F_OUT><<<cdiv((long)E_EDGES * (F_OUT / 4), T), T, 0, stream>>>(
        H2, row, col, dinv, out, E_EDGES);
    finalize_kernel<F_OUT, false><<<cdiv((long)N_NODES * F_OUT, T), T, 0, stream>>>(
        out, H2, dinv, b2, N_NODES);
}
